// Block_72980084294384
// MI455X (gfx1250) — matrix operand order repaired and checked by reading, not compiled, not run on hardware
//
#include <hip/hip_runtime.h>
#include <hip/hip_bf16.h>
#include <math.h>
#include <stdint.h>

// Problem dims (fixed by the reference)
#define BQ 4
#define TQ 2048
#define CQ 2048
#define FQ 8192
#define MQ (BQ * TQ)   // 8192 tokens

typedef __attribute__((ext_vector_type(16))) __bf16 bf16x16;
typedef __attribute__((ext_vector_type(8)))  float  f32x8;
typedef __attribute__((ext_vector_type(4)))  unsigned int u32x4;

union FragU { bf16x16 f; u32x4 q[2]; };

// Async global->LDS path (CDNA5). s_wait_asynccnt builtin is probe-confirmed;
// the copy itself is inline asm (per-lane 16B: vdst = LDS addr, vaddr pair).
#if defined(__gfx1250__) && __has_builtin(__builtin_amdgcn_s_wait_asynccnt)
#define ASYNC_LDS 1
#else
#define ASYNC_LDS 0
#endif

__device__ __forceinline__ void async_copy16(const __bf16* g, char* l)
{
#if ASYNC_LDS
    const unsigned lds_off =
        (unsigned)(uintptr_t)(__attribute__((address_space(3))) char*)l;
    const unsigned long long ga = (unsigned long long)(uintptr_t)g;
    asm volatile("global_load_async_to_lds_b128 %0, %1, off"
                 :: "v"(lds_off), "v"(ga) : "memory");
#else
    *(u32x4*)l = *(const u32x4*)g;   // global_load_b128 + ds_store_b128
#endif
}

__device__ __forceinline__ void async_wait()
{
#if ASYNC_LDS
    __builtin_amdgcn_s_wait_asynccnt(0);
#endif
}

// ---------------------------------------------------------------------------
// LDS tile geometry: A tile 128x32 bf16, B tile 256x32 bf16, rows padded to
// 80B (64B data + 16B pad) so 16 consecutive rows map to distinct banks.
// ---------------------------------------------------------------------------
#define LDS_STR   80
#define A_BYTES   (128 * LDS_STR)          // 10240
#define B_BYTES   (256 * LDS_STR)          // 20480
#define BUF_BYTES (A_BYTES + B_BYTES)      // 30720

__device__ __forceinline__ void stage_tile(
    const __bf16* __restrict__ A, const __bf16* __restrict__ W,
    char* buf, int blkM, int blkN, size_t Kdim, int k0, int tid)
{
    char* ldsA = buf;
    char* ldsB = buf + A_BYTES;
    // A: 128 rows x 4 chunks(16B) = 512 chunks, 2 per thread
#pragma unroll
    for (int s = 0; s < 2; ++s) {
        const int q = tid + s * 256;
        const int row = q >> 2, ch = q & 3;
        async_copy16(A + (size_t)(blkM + row) * Kdim + k0 + ch * 8,
                     ldsA + row * LDS_STR + ch * 16);
    }
    // B: 256 rows x 4 chunks = 1024 chunks, 4 per thread
#pragma unroll
    for (int s = 0; s < 4; ++s) {
        const int q = tid + s * 256;
        const int row = q >> 2, ch = q & 3;
        async_copy16(W + (size_t)(blkN + row) * Kdim + k0 + ch * 8,
                     ldsB + row * LDS_STR + ch * 16);
    }
}

// Fragment reads from LDS per CDNA5 ISA 7.12.2 (wave32)
// A 16x32: lane half 0 holds K{0..7,16..23}, half 1 holds K{8..15,24..31}
__device__ __forceinline__ bf16x16 lds_fragA(const char* ldsA, int row, int laneHi)
{
    const char* p = ldsA + row * LDS_STR + laneHi * 16;
    FragU u;
    u.q[0] = *(const u32x4*)p;
    u.q[1] = *(const u32x4*)(p + 32);
    return u.f;
}
// B 32x16: lane half 0 holds K{0..7,16..23}, half 1 holds K{8..15,24..31}
__device__ __forceinline__ bf16x16 lds_fragB(const char* ldsB, int row, int laneHi)
{
    const char* p = ldsB + row * LDS_STR + laneHi * 16;
    FragU u;
    u.q[0] = *(const u32x4*)p;
    u.q[1] = *(const u32x4*)(p + 32);
    return u.f;
}

// ---------------------------------------------------------------------------
// GEMM: D[M,N] = A[M,K] * W[N,K]^T  (bf16 in, f32 accum via WMMA)
// Block tile 128x256, 8 waves (2x4), each wave 64x64 = 4x4 WMMA tiles.
// Double-buffered async global->LDS staging, one barrier per K step.
// mode 0: f32   1: sigmoid   2: bf16(relu^2) -> outH   3: f32 + resid
// ---------------------------------------------------------------------------
__global__ __launch_bounds__(256) void gemm_bf16_wmma(
    const __bf16* __restrict__ A, const __bf16* __restrict__ W,
    float* __restrict__ outF, __bf16* __restrict__ outH,
    const float* __restrict__ resid,
    int Ndim, int Kdim, int mode)
{
    __shared__ alignas(16) char lds[2 * BUF_BYTES];
    const int tid     = threadIdx.x;
    const int lane    = tid & 31;
    const int wid     = tid >> 5;
    const int laneRow = lane & 15;
    const int laneHi  = lane >> 4;
    const int blkM    = blockIdx.x * 128;
    const int blkN    = blockIdx.y * 256;
    const int mW      = (wid & 1) * 64;    // wave row within block tile
    const int nW      = (wid >> 1) * 64;   // wave col within block tile

    f32x8 acc[4][4];
#pragma unroll
    for (int i = 0; i < 4; ++i)
#pragma unroll
        for (int j = 0; j < 4; ++j)
#pragma unroll
            for (int r = 0; r < 8; ++r) acc[i][j][r] = 0.0f;

    const int nk = Kdim >> 5;
    stage_tile(A, W, lds, blkM, blkN, (size_t)Kdim, 0, tid);

    for (int it = 0; it < nk; ++it) {
        async_wait();           // my async writes into current buffer done
        __syncthreads();        // everyone's writes done, prev reads done
        const char* cur = lds + (size_t)(it & 1) * BUF_BYTES;
        if (it + 1 < nk)
            stage_tile(A, W, lds + (size_t)((it + 1) & 1) * BUF_BYTES,
                       blkM, blkN, (size_t)Kdim, (it + 1) << 5, tid);
        const char* ldsA = cur;
        const char* ldsB = cur + A_BYTES;

        bf16x16 af[4], bfr[4];
#pragma unroll
        for (int i = 0; i < 4; ++i)
            af[i] = lds_fragA(ldsA, mW + i * 16 + laneRow, laneHi);
#pragma unroll
        for (int j = 0; j < 4; ++j)
            bfr[j] = lds_fragB(ldsB, nW + j * 16 + laneRow, laneHi);
#pragma unroll
        for (int i = 0; i < 4; ++i)
#pragma unroll
            for (int j = 0; j < 4; ++j)
                acc[i][j] = __builtin_amdgcn_wmma_f32_16x16x32_bf16(
                    false, af[i], false, bfr[j], (short)0, acc[i][j], false, false);
    }

    // Epilogue: lane holds column n, rows m = laneHi*8 + r
#pragma unroll
    for (int i = 0; i < 4; ++i) {
#pragma unroll
        for (int j = 0; j < 4; ++j) {
            const int n = blkN + nW + j * 16 + laneRow;
#pragma unroll
            for (int r = 0; r < 8; ++r) {
                const int m = blkM + mW + i * 16 + laneHi * 8 + r;
                const size_t o = (size_t)m * Ndim + n;
                const float v = acc[i][j][r];
                if (mode == 0) {
                    outF[o] = v;
                } else if (mode == 1) {
                    outF[o] = 1.0f / (1.0f + __expf(-v));
                } else if (mode == 2) {
                    const float t = fmaxf(v, 0.0f);
                    outH[o] = (__bf16)(t * t);
                } else {
                    outF[o] = v + resid[o];
                }
            }
        }
    }
}

// ---------------------------------------------------------------------------
// LayerNorm over last dim (C), one block per (b,t) row.
// ---------------------------------------------------------------------------
__global__ __launch_bounds__(256) void ln_kernel(
    const float* __restrict__ x, const float* __restrict__ w,
    const float* __restrict__ b, float* __restrict__ out)
{
    __shared__ float sred[256];
    const int tid = threadIdx.x;
    const size_t row = blockIdx.x;
    const float* xr = x + row * CQ;

    float s = 0.0f;
    for (int c = tid; c < CQ; c += 256) s += xr[c];
    sred[tid] = s; __syncthreads();
    for (int st = 128; st > 0; st >>= 1) {
        if (tid < st) sred[tid] += sred[tid + st];
        __syncthreads();
    }
    const float mean = sred[0] * (1.0f / CQ);
    __syncthreads();

    float vs = 0.0f;
    for (int c = tid; c < CQ; c += 256) {
        const float d = xr[c] - mean;
        vs += d * d;
    }
    sred[tid] = vs; __syncthreads();
    for (int st = 128; st > 0; st >>= 1) {
        if (tid < st) sred[tid] += sred[tid + st];
        __syncthreads();
    }
    const float rstd = rsqrtf(sred[0] * (1.0f / CQ) + 1e-5f);

    float* orow = out + row * CQ;
    for (int c = tid; c < CQ; c += 256)
        orow[c] = (xr[c] - mean) * rstd * w[c] + b[c];
}

// Time-shift mixing (3 outputs, bf16 for WMMA)
__global__ __launch_bounds__(256) void mix3_kernel(
    const float* __restrict__ x1, const float* __restrict__ shift,
    const float* __restrict__ mk, const float* __restrict__ mv,
    const float* __restrict__ mr,
    __bf16* __restrict__ xk, __bf16* __restrict__ xv, __bf16* __restrict__ xr)
{
    const size_t i = (size_t)blockIdx.x * 256 + threadIdx.x;
    const int c = (int)(i % CQ);
    const size_t bt = i / CQ;
    const int t = (int)(bt % TQ);
    const int b = (int)(bt / TQ);
    const float cur  = x1[i];
    const float prev = (t == 0) ? shift[(size_t)b * CQ + c] : x1[i - CQ];
    float a = mk[c]; xk[i] = (__bf16)(cur * a + prev * (1.0f - a));
    a = mv[c];       xv[i] = (__bf16)(cur * a + prev * (1.0f - a));
    a = mr[c];       xr[i] = (__bf16)(cur * a + prev * (1.0f - a));
}

// Time-shift mixing (2 outputs) for FFN
__global__ __launch_bounds__(256) void mix2_kernel(
    const float* __restrict__ x2, const float* __restrict__ shift,
    const float* __restrict__ mk, const float* __restrict__ mr,
    __bf16* __restrict__ xk, __bf16* __restrict__ xr)
{
    const size_t i = (size_t)blockIdx.x * 256 + threadIdx.x;
    const int c = (int)(i % CQ);
    const size_t bt = i / CQ;
    const int t = (int)(bt % TQ);
    const int b = (int)(bt / TQ);
    const float cur  = x2[i];
    const float prev = (t == 0) ? shift[(size_t)b * CQ + c] : x2[i - CQ];
    float a = mk[c]; xk[i] = (__bf16)(cur * a + prev * (1.0f - a));
    a = mr[c];       xr[i] = (__bf16)(cur * a + prev * (1.0f - a));
}

// WKV recurrence: one lane per (b, c) channel, serial loop over T.
__global__ __launch_bounds__(256) void wkv_kernel(
    const float* __restrict__ k, const float* __restrict__ v,
    const float* __restrict__ time_decay, const float* __restrict__ time_first,
    const float* __restrict__ state0, float* __restrict__ y,
    float* __restrict__ stateOut)
{
    const int idx = blockIdx.x * 256 + threadIdx.x;   // 0 .. B*C-1
    const int b = idx / CQ;
    const int c = idx % CQ;
    const float u = time_first[c];
    const float w = -__expf(time_decay[c]);
    float aa = state0[(size_t)idx * 3 + 0];
    float bb = state0[(size_t)idx * 3 + 1];
    float pp = state0[(size_t)idx * 3 + 2];
    const float* kp = k + (size_t)b * TQ * CQ + c;
    const float* vp = v + (size_t)b * TQ * CQ + c;
    float* yp = y + (size_t)b * TQ * CQ + c;
    for (int t = 0; t < TQ; ++t) {
        const float kk = kp[(size_t)t * CQ];
        const float vv = vp[(size_t)t * CQ];
        const float ww = u + kk;
        const float p  = fmaxf(pp, ww);
        const float e1 = __expf(pp - p);
        const float e2 = __expf(ww - p);
        yp[(size_t)t * CQ] = (e1 * aa + e2 * vv) / (e1 * bb + e2);
        const float ww2 = pp + w;
        const float p2  = fmaxf(ww2, kk);
        const float e1b = __expf(ww2 - p2);
        const float e2b = __expf(kk - p2);
        aa = e1b * aa + e2b * vv;
        bb = e1b * bb + e2b;
        pp = p2;
    }
    stateOut[(size_t)idx * 3 + 0] = aa;
    stateOut[(size_t)idx * 3 + 1] = bb;
    stateOut[(size_t)idx * 3 + 2] = pp;
}

// sry = bf16(sr * y)
__global__ __launch_bounds__(256) void sry_kernel(
    const float* __restrict__ sr, const float* __restrict__ y,
    __bf16* __restrict__ sry)
{
    const size_t i = (size_t)blockIdx.x * 256 + threadIdx.x;
    sry[i] = (__bf16)(sr[i] * y[i]);
}

// x += fr * kv (fr already sigmoid'ed)
__global__ __launch_bounds__(256) void final_kernel(
    float* __restrict__ xo, const float* __restrict__ fr,
    const float* __restrict__ kv)
{
    const size_t i = (size_t)blockIdx.x * 256 + threadIdx.x;
    xo[i] = xo[i] + fr[i] * kv[i];
}

// copy last timestep row: out[b, c] = src[b, T-1, c]
__global__ __launch_bounds__(256) void copy_last_kernel(
    const float* __restrict__ src, float* __restrict__ out)
{
    const int i = blockIdx.x * 256 + threadIdx.x;    // 0..B*C-1
    const int b = i / CQ;
    const int c = i % CQ;
    out[i] = src[((size_t)b * TQ + (TQ - 1)) * CQ + c];
}

// fp32 -> bf16
__global__ __launch_bounds__(256) void cvt_kernel(
    const float* __restrict__ f, __bf16* __restrict__ h)
{
    const size_t i = (size_t)blockIdx.x * 256 + threadIdx.x;
    h[i] = (__bf16)f[i];
}

// ---------------------------------------------------------------------------
// Workspace layout (bump offsets, bytes).  Peak ~478 MB with reuse.
// ---------------------------------------------------------------------------
static constexpr size_t SZ_W   = (size_t)CQ * CQ * 2;
static constexpr size_t SZ_FW  = (size_t)CQ * FQ * 2;
static constexpr size_t SZ_AH  = (size_t)MQ * CQ * 2;
static constexpr size_t SZ_AF  = (size_t)MQ * CQ * 4;

static constexpr size_t oWkH  = 0;
static constexpr size_t oWvH  = oWkH + SZ_W;
static constexpr size_t oWrH  = oWvH + SZ_W;
static constexpr size_t oWoH  = oWrH + SZ_W;
static constexpr size_t ofWkH = oWoH + SZ_W;
static constexpr size_t ofWvH = ofWkH + SZ_FW;
static constexpr size_t ofWrH = ofWvH + SZ_FW;
static constexpr size_t oX1   = ofWrH + SZ_W;   // also y, kv
static constexpr size_t oXkH  = oX1 + SZ_AF;    // also sryH
static constexpr size_t oXvH  = oXkH + SZ_AH;   // also fxkH
static constexpr size_t oXrH  = oXvH + SZ_AH;   // also fxrH
static constexpr size_t oK    = oXrH + SZ_AH;   // also x2, fr
static constexpr size_t oV    = oK + SZ_AF;     // fkH spans oV..oV+2*SZ_AF
static constexpr size_t oSr   = oV + SZ_AF;

// d_out layout (f32 elements): x | att_shift | wkv | ffn_shift
static constexpr size_t OUT_X   = 0;
static constexpr size_t OUT_ATT = (size_t)MQ * CQ;
static constexpr size_t OUT_WKV = OUT_ATT + (size_t)BQ * CQ;
static constexpr size_t OUT_FFN = OUT_WKV + (size_t)BQ * CQ * 3;

extern "C" void kernel_launch(void* const* d_in, const int* in_sizes, int n_in,
                              void* d_out, int out_size, void* d_ws, size_t ws_size,
                              hipStream_t stream)
{
    const float* x      = (const float*)d_in[0];
    const float* attsh  = (const float*)d_in[1];
    const float* wkvst  = (const float*)d_in[2];
    const float* ffnsh  = (const float*)d_in[3];
    const float* ln1w   = (const float*)d_in[4];
    const float* ln1b   = (const float*)d_in[5];
    const float* ln2w   = (const float*)d_in[6];
    const float* ln2b   = (const float*)d_in[7];
    const float* tdecay = (const float*)d_in[8];
    const float* tfirst = (const float*)d_in[9];
    const float* tmk    = (const float*)d_in[10];
    const float* tmv    = (const float*)d_in[11];
    const float* tmr    = (const float*)d_in[12];
    const float* Wk     = (const float*)d_in[13];
    const float* Wv     = (const float*)d_in[14];
    const float* Wr     = (const float*)d_in[15];
    const float* Wo     = (const float*)d_in[16];
    const float* ftmk   = (const float*)d_in[17];
    const float* ftmr   = (const float*)d_in[18];
    const float* fWk    = (const float*)d_in[19];
    const float* fWv    = (const float*)d_in[20];
    const float* fWr    = (const float*)d_in[21];

    char* ws = (char*)d_ws;
    __bf16* WkH  = (__bf16*)(ws + oWkH);
    __bf16* WvH  = (__bf16*)(ws + oWvH);
    __bf16* WrH  = (__bf16*)(ws + oWrH);
    __bf16* WoH  = (__bf16*)(ws + oWoH);
    __bf16* fWkH = (__bf16*)(ws + ofWkH);
    __bf16* fWvH = (__bf16*)(ws + ofWvH);
    __bf16* fWrH = (__bf16*)(ws + ofWrH);
    float*  x1   = (float*)(ws + oX1);
    float*  ybuf = (float*)(ws + oX1);
    float*  kv   = (float*)(ws + oX1);
    __bf16* xkH  = (__bf16*)(ws + oXkH);
    __bf16* sryH = (__bf16*)(ws + oXkH);
    __bf16* xvH  = (__bf16*)(ws + oXvH);
    __bf16* fxkH = (__bf16*)(ws + oXvH);
    __bf16* xrH  = (__bf16*)(ws + oXrH);
    __bf16* fxrH = (__bf16*)(ws + oXrH);
    float*  kbuf = (float*)(ws + oK);
    float*  x2   = (float*)(ws + oK);
    float*  fr   = (float*)(ws + oK);
    float*  vbuf = (float*)(ws + oV);
    __bf16* fkH  = (__bf16*)(ws + oV);
    float*  srb  = (float*)(ws + oSr);

    float* out_x   = (float*)d_out + OUT_X;
    float* out_att = (float*)d_out + OUT_ATT;
    float* out_wkv = (float*)d_out + OUT_WKV;
    float* out_ffn = (float*)d_out + OUT_FFN;

    const int EW = (int)((size_t)MQ * CQ / 256);
    const int CC = (int)((size_t)CQ * CQ / 256);
    const int CF = (int)((size_t)CQ * FQ / 256);
    const int BC = (BQ * CQ) / 256;

    // 1) weights -> bf16
    cvt_kernel<<<CC, 256, 0, stream>>>(Wk, WkH);
    cvt_kernel<<<CC, 256, 0, stream>>>(Wv, WvH);
    cvt_kernel<<<CC, 256, 0, stream>>>(Wr, WrH);
    cvt_kernel<<<CC, 256, 0, stream>>>(Wo, WoH);
    cvt_kernel<<<CF, 256, 0, stream>>>(fWk, fWkH);
    cvt_kernel<<<CF, 256, 0, stream>>>(fWv, fWvH);
    cvt_kernel<<<CC, 256, 0, stream>>>(fWr, fWrH);

    // 2) LN1, shift outputs, mixing
    ln_kernel<<<MQ, 256, 0, stream>>>(x, ln1w, ln1b, x1);
    copy_last_kernel<<<BC, 256, 0, stream>>>(x1, out_att);
    mix3_kernel<<<EW, 256, 0, stream>>>(x1, attsh, tmk, tmv, tmr, xkH, xvH, xrH);

    // 3) k / v / sigmoid(r) projections (WMMA, 128x256 block tiles)
    dim3 gC(MQ / 128, CQ / 256);
    gemm_bf16_wmma<<<gC, 256, 0, stream>>>(xkH, WkH, kbuf, nullptr, nullptr, CQ, CQ, 0);
    gemm_bf16_wmma<<<gC, 256, 0, stream>>>(xvH, WvH, vbuf, nullptr, nullptr, CQ, CQ, 0);
    gemm_bf16_wmma<<<gC, 256, 0, stream>>>(xrH, WrH, srb,  nullptr, nullptr, CQ, CQ, 1);

    // 4) WKV recurrence (serial in T, parallel over B*C channels)
    wkv_kernel<<<BC, 256, 0, stream>>>(kbuf, vbuf, tdecay, tfirst, wkvst, ybuf, out_wkv);

    // 5) att_out = (sr*y) @ Wo^T + x   -> d_out x region
    sry_kernel<<<EW, 256, 0, stream>>>(srb, ybuf, sryH);
    gemm_bf16_wmma<<<gC, 256, 0, stream>>>(sryH, WoH, out_x, nullptr, x, CQ, CQ, 3);

    // 6) FFN path
    ln_kernel<<<MQ, 256, 0, stream>>>(out_x, ln2w, ln2b, x2);
    copy_last_kernel<<<BC, 256, 0, stream>>>(x2, out_ffn);
    mix2_kernel<<<EW, 256, 0, stream>>>(x2, ffnsh, ftmk, ftmr, fxkH, fxrH);

    dim3 gF(MQ / 128, FQ / 256);
    gemm_bf16_wmma<<<gF, 256, 0, stream>>>(fxkH, fWkH, nullptr, fkH, nullptr, FQ, CQ, 2);
    gemm_bf16_wmma<<<gC, 256, 0, stream>>>(fkH, fWvH, kv, nullptr, nullptr, CQ, FQ, 0);
    gemm_bf16_wmma<<<gC, 256, 0, stream>>>(fxrH, fWrH, fr, nullptr, nullptr, CQ, CQ, 1);

    final_kernel<<<EW, 256, 0, stream>>>(out_x, fr, kv);
}
